// SelfAttention_39324720562980
// MI455X (gfx1250) — matrix operand order repaired and checked by reading, not compiled, not run on hardware
//
#include <hip/hip_runtime.h>
#include <hip/hip_bf16.h>

// ---------------------------------------------------------------------------
// Problem constants
// ---------------------------------------------------------------------------
#define BATCH   2
#define SLEN    4096
#define DM      512
#define DH      512
#define NHEAD   8
#define HDIM    64
#define RANK    4
#define MROWS   (BATCH * SLEN)      // 8192
#define KQ_LD   (2 * DH)            // 1024 (K | Q packed)
#define LORA_SCALE 0.25f
// softmax done in exp2 domain: scores pre-scaled by (1/sqrt(64)) * log2(e)
#define SCALE_LOG2E (0.125f * 1.44269504088896f)

typedef __attribute__((ext_vector_type(16))) __bf16 v16bf;
typedef __attribute__((ext_vector_type(8)))  float  v8f;
typedef __attribute__((ext_vector_type(4)))  float  f32x4;
typedef __attribute__((ext_vector_type(4)))  unsigned int u32x4;
typedef __attribute__((ext_vector_type(8)))  int    i32x8;
typedef __attribute__((ext_vector_type(4)))  int    i32x4;

// ---------------------------------------------------------------------------
// CDNA5 Tensor Data Mover support (compile-probed, with manual fallback)
// ---------------------------------------------------------------------------
#if defined(__HIP_DEVICE_COMPILE__) && __has_builtin(__builtin_amdgcn_tensor_load_to_lds)
#define HAVE_TDM 1
#else
#define HAVE_TDM 0
#endif

#if defined(__HIP_DEVICE_COMPILE__) && __has_builtin(__builtin_amdgcn_s_wait_tensorcnt)
#define WAIT_TENSORCNT(n) __builtin_amdgcn_s_wait_tensorcnt(n)
#elif defined(__HIP_DEVICE_COMPILE__)
#define WAIT_TENSORCNT(n) asm volatile("s_wait_tensorcnt %0" ::"i"(n) : "memory")
#else
#define WAIT_TENSORCNT(n)
#endif

#if HAVE_TDM
// DMA a 2-D bf16 tile [tile_y rows x tile_x elems] from global (row stride
// stride_x elems) into LDS at byte offset lds_off.  D# packed per ISA §8.3/8.4:
// group0 = {count/type/addr}, group1 = {data_size, dims, tile dims, strides}.
__device__ __forceinline__ void tdm_load_tile_bf16(
    const void* gptr, unsigned lds_off, unsigned tile_x, unsigned tile_y,
    unsigned stride_x, unsigned tensor_x, unsigned tensor_y) {
    unsigned long long ga = (unsigned long long)(uintptr_t)gptr;
    u32x4 g0 = {1u,                                      // count=1, user desc
                lds_off,                                 // lds_addr
                (unsigned)(ga & 0xffffffffu),            // global_addr lo
                (unsigned)((ga >> 32) & 0x1ffffffu) | (2u << 30)};  // hi|type=2
    unsigned d0 = 1u << 16;                              // data_size=1 -> 2B
    unsigned d1 = (tensor_x & 0xffffu) << 16;            // tensor_dim0 lo16
    unsigned d2 = (tensor_x >> 16) | ((tensor_y & 0xffffu) << 16);
    unsigned d3 = (tensor_y >> 16) | (tile_x << 16);     // tile_dim0
    unsigned d4 = tile_y;                                // tile_dim1, dim2=0
    unsigned d5 = stride_x;                              // dim0_stride lo32
    unsigned d6 = 0, d7 = 0;                             // stride hi, dim1_stride
    i32x8 g1 = {(int)d0, (int)d1, (int)d2, (int)d3,
                (int)d4, (int)d5, (int)d6, (int)d7};
    i32x4 gz = {0, 0, 0, 0};
#if __has_include(<hip/amd_detail/amd_gfx1250_TDM.h>)
    i32x8 gz8 = {0, 0, 0, 0, 0, 0, 0, 0};
    __builtin_amdgcn_tensor_load_to_lds(g0, g1, gz, gz, gz8, 0);
#else
    __builtin_amdgcn_tensor_load_to_lds(g0, g1, gz, gz, 0);
#endif
}
#endif

__device__ __forceinline__ v8f zero8() {
    v8f z = {0.f, 0.f, 0.f, 0.f, 0.f, 0.f, 0.f, 0.f};
    return z;
}

// A-matrix fragment, 16x32 bf16 (M x K), row-major source, wave32.
// lane<16: M=lane,    K = kbase + {0..7, 16..23}
// lane>=16:M=lane-16, K = kbase + {8..15, 24..31}
__device__ __forceinline__ v16bf load_a_frag(const __bf16* base, int ld,
                                             int mrow, int kbase, int lane) {
    const int m  = mrow + (lane & 15);
    const int hf = lane >> 4;
    const __bf16* p = base + (size_t)m * ld + kbase + hf * 8;
    union { f32x4 f4[2]; v16bf v; } u;
    u.f4[0] = *reinterpret_cast<const f32x4*>(p);
    u.f4[1] = *reinterpret_cast<const f32x4*>(p + 16);
    return u.v;
}

// B-matrix fragment, 32x16 bf16 (K x N), loaded from row-major [N,K] source.
// lane<16: N=lane,    K = kbase + {0..7, 16..23}
// lane>=16:N=lane-16, K = kbase + {8..15, 24..31}
__device__ __forceinline__ v16bf load_b_frag(const __bf16* base, int ld,
                                             int nrow, int kbase, int lane) {
    const int n  = nrow + (lane & 15);
    const int hf = lane >> 4;
    const __bf16* p = base + (size_t)n * ld + kbase + hf * 8;
    union { f32x4 f4[2]; v16bf v; } u;
    u.f4[0] = *reinterpret_cast<const f32x4*>(p);
    u.f4[1] = *reinterpret_cast<const f32x4*>(p + 16);
    return u.v;
}

// ---------------------------------------------------------------------------
// Prep kernels
// ---------------------------------------------------------------------------
__global__ void cvt_f32_to_bf16(const float* __restrict__ src,
                                __bf16* __restrict__ dst, int n) {
    int i = blockIdx.x * blockDim.x + threadIdx.x;
    if (i < n) dst[i] = (__bf16)src[i];
}

// Packed effective weight: rows [0,512)=Wk, [512,1024)=Wq+0.25*Bq@Aq,
// [1024,1536)=Wv+0.25*Bv@Av.  All [N,K] row-major, K=DM.
__global__ void build_w_all(const float* __restrict__ Wk,
                            const float* __restrict__ Wq,
                            const float* __restrict__ Wv,
                            const float* __restrict__ Aq,
                            const float* __restrict__ Bq,
                            const float* __restrict__ Av,
                            const float* __restrict__ Bv,
                            __bf16* __restrict__ wall) {
    int i = blockIdx.x * blockDim.x + threadIdx.x;
    if (i >= 3 * DH * DM) return;
    int n = i >> 9;        // /512
    int k = i & (DM - 1);
    float w;
    if (n < DH) {
        w = Wk[n * DM + k];
    } else if (n < 2 * DH) {
        int nn = n - DH;
        float l = 0.f;
#pragma unroll
        for (int r = 0; r < RANK; r++) l += Bq[nn * RANK + r] * Aq[r * DM + k];
        w = Wq[nn * DM + k] + LORA_SCALE * l;
    } else {
        int nn = n - 2 * DH;
        float l = 0.f;
#pragma unroll
        for (int r = 0; r < RANK; r++) l += Bv[nn * RANK + r] * Av[r * DM + k];
        w = Wv[nn * DM + k] + LORA_SCALE * l;
    }
    wall[i] = (__bf16)w;
}

// ---------------------------------------------------------------------------
// Tiled bf16 WMMA GEMM:  C[M,N] = A[M,K] * W[N,K]^T  (both row-major)
// Block tile 128x128, 256 threads (8 waves, 4x2), wave tile 32x64 (2x4 WMMA).
// TDM double-buffered LDS staging (A and W tiles), TENSORcnt synchronized.
// MODE: 0 = bf16 row-major store, 1 = f32 + bias store,
//       2 = bf16 transposed V store into vt[b*H+h][d][s] (for flash attn).
// LDS arena: A[2][128][32] @0, W[2][128][32] @16384   (32 KB dynamic)
// ---------------------------------------------------------------------------
#define GEMM_ABUF  (128 * 32 * 2)   // 8192 B per buffer
#define GEMM_WOFF  (2 * GEMM_ABUF)  // 16384
#define GEMM_LDS   (4 * GEMM_ABUF)  // 32768

template <int MODE>
__global__ __launch_bounds__(256) void gemm_bf16_wmma(
    const __bf16* __restrict__ A, const __bf16* __restrict__ W,
    __bf16* __restrict__ outB, float* __restrict__ outF,
    const float* __restrict__ bias, int K, int ldc) {
    extern __shared__ char smem[];

    const int tid  = threadIdx.x;
    const int lane = tid & 31;
    const int wid  = tid >> 5;
    const int bm   = blockIdx.x * 128;
    const int bn   = blockIdx.y * 128;
    const int wm   = (wid >> 1) * 32;   // 0,32,64,96
    const int wn   = (wid & 1) * 64;    // 0,64

    v8f acc[2][4];
#pragma unroll
    for (int i = 0; i < 2; i++)
#pragma unroll
        for (int j = 0; j < 4; j++) acc[i][j] = zero8();

    const int nk = K / 32;
#if HAVE_TDM
    if (wid == 0) {
        tdm_load_tile_bf16(A + (size_t)bm * K, 0, 32, 128, K, K, 1u << 20);
        tdm_load_tile_bf16(W + (size_t)bn * K, GEMM_WOFF, 32, 128, K, K,
                           1u << 20);
    }
#endif
    for (int i = 0; i < nk; i++) {
        const int cur = i & 1;
        __syncthreads();  // A: everyone done reading buffer cur^1
#if HAVE_TDM
        if (wid == 0) {
            if (i + 1 < nk) {
                const int nxt = cur ^ 1;
                tdm_load_tile_bf16(A + (size_t)bm * K + (i + 1) * 32,
                                   nxt * GEMM_ABUF, 32, 128, K, K, 1u << 20);
                tdm_load_tile_bf16(W + (size_t)bn * K + (i + 1) * 32,
                                   GEMM_WOFF + nxt * GEMM_ABUF, 32, 128, K, K,
                                   1u << 20);
                WAIT_TENSORCNT(2);  // tile i landed; i+1 still in flight
            } else {
                WAIT_TENSORCNT(0);
            }
        }
#else
        {  // manual staging of buffer cur
            __bf16* la = (__bf16*)(smem + cur * GEMM_ABUF);
            __bf16* lw = (__bf16*)(smem + GEMM_WOFF + cur * GEMM_ABUF);
#pragma unroll
            for (int c = 0; c < 2; c++) {
                int ch  = tid * 2 + c;
                int row = ch >> 2;
                int cc  = (ch & 3) * 8;
                *reinterpret_cast<f32x4*>(&la[row * 32 + cc]) =
                    *reinterpret_cast<const f32x4*>(A + (size_t)(bm + row) * K +
                                                    i * 32 + cc);
                *reinterpret_cast<f32x4*>(&lw[row * 32 + cc]) =
                    *reinterpret_cast<const f32x4*>(W + (size_t)(bn + row) * K +
                                                    i * 32 + cc);
            }
        }
#endif
        __syncthreads();  // B: buffer cur ready
        const __bf16* la = (const __bf16*)(smem + cur * GEMM_ABUF);
        const __bf16* lw = (const __bf16*)(smem + GEMM_WOFF + cur * GEMM_ABUF);
        v16bf af[2], wf[4];
#pragma unroll
        for (int ii = 0; ii < 2; ii++)
            af[ii] = load_a_frag(la, 32, wm + ii * 16, 0, lane);
#pragma unroll
        for (int jj = 0; jj < 4; jj++)
            wf[jj] = load_b_frag(lw, 32, wn + jj * 16, 0, lane);
#pragma unroll
        for (int ii = 0; ii < 2; ii++)
#pragma unroll
            for (int jj = 0; jj < 4; jj++)
                acc[ii][jj] = __builtin_amdgcn_wmma_f32_16x16x32_bf16(
                    false, af[ii], false, wf[jj], (short)0, acc[ii][jj], false,
                    false);
    }

    // Epilogue. C layout: VGPR r -> row r + 8*(lane/16), col = lane%16.
    const int ln = lane & 15;
    const int hf = lane >> 4;
#pragma unroll
    for (int i = 0; i < 2; i++)
#pragma unroll
        for (int j = 0; j < 4; j++)
#pragma unroll
            for (int r = 0; r < 8; r++) {
                int row = bm + wm + i * 16 + r + 8 * hf;
                int col = bn + wn + j * 16 + ln;
                float v = acc[i][j][r];
                if constexpr (MODE == 1) {
                    outF[(size_t)row * ldc + col] = v + bias[col];
                } else if constexpr (MODE == 2) {
                    // row = b*SLEN + s, col = h*HDIM + d; store V^T:
                    // vt[((b*NHEAD+h)*HDIM + d)*SLEN + s]
                    int bb = row >> 12, s = row & (SLEN - 1);
                    int hh = col >> 6, d = col & (HDIM - 1);
                    outB[((size_t)((((bb << 3) + hh) << 6) | d) << 12) + s] =
                        (__bf16)v;
                } else {
                    outB[(size_t)row * ldc + col] = (__bf16)v;
                }
            }
}

// ---------------------------------------------------------------------------
// Flash attention (bf16 WMMA, f32 accum, online softmax in exp2 domain).
// Grid: x = S/128 query blocks, y = B*H. Block: 256 threads = 8 waves,
// each wave owns 16 query rows (128 queries share each K/V tile -> 2x the
// L2 reuse of the 64-query version). Key blocks of 64; BOTH the K tile
// [key][d] and the pre-transposed V^T tile [d][key] are TDM double-buffered;
// only the 64-entry mask is thread-staged.
// LDS arena: K[2][64][64] @0, Vt[2][64][64] @16384, P[8][16][64] @32768,
//            mask[64] @49152   (~48 KB dynamic)
// ---------------------------------------------------------------------------
#define FA_TBUF   (64 * 64 * 2)          // 8192 B
#define FA_VTOFF  (2 * FA_TBUF)          // 16384
#define FA_POFF   (4 * FA_TBUF)          // 32768
#define FA_MOFF   (FA_POFF + 8 * 16 * 64 * 2)  // 49152
#define FA_LDS    (FA_MOFF + 64 * 4)     // 49408

__global__ __launch_bounds__(256) void flash_attn_wmma(
    const __bf16* __restrict__ kq, const __bf16* __restrict__ vt,
    const int* __restrict__ amask, __bf16* __restrict__ ctx) {
    extern __shared__ char smem[];
    __bf16* ldsP = (__bf16*)(smem + FA_POFF);
    int*    ldsM = (int*)(smem + FA_MOFF);

    const int tid  = threadIdx.x;
    const int lane = tid & 31;
    const int wid  = tid >> 5;          // 0..7
    const int ln   = lane & 15;
    const int hf   = lane >> 4;

    const int bh = blockIdx.y;
    const int b  = bh >> 3;
    const int h  = bh & 7;
    const int qrow0 = blockIdx.x * 128 + wid * 16;

    const __bf16* kbasep  = kq + (size_t)(b * SLEN) * KQ_LD + 0 * DH + h * HDIM;
    const __bf16* qbasep  = kq + (size_t)(b * SLEN) * KQ_LD + 1 * DH + h * HDIM;
    const __bf16* vtbasep = vt + ((size_t)(b * NHEAD + h) * HDIM) * SLEN;

    // Q fragments for this wave's 16 rows (HD=64 -> two K-steps), loaded once.
    v16bf qf[2];
    qf[0] = load_a_frag(qbasep, KQ_LD, qrow0, 0, lane);
    qf[1] = load_a_frag(qbasep, KQ_LD, qrow0, 32, lane);

    v8f O[4];
#pragma unroll
    for (int t = 0; t < 4; t++) O[t] = zero8();
    float mrow[8], lrow[8];
#pragma unroll
    for (int r = 0; r < 8; r++) { mrow[r] = -INFINITY; lrow[r] = 0.f; }

    const int nkb = SLEN / 64;
#if HAVE_TDM
    if (wid == 0) {
        tdm_load_tile_bf16(kbasep, 0, 64, 64, KQ_LD, KQ_LD, 1u << 20);
        tdm_load_tile_bf16(vtbasep, FA_VTOFF, 64, 64, SLEN, SLEN, 1u << 20);
    }
#endif
    for (int i = 0; i < nkb; i++) {
        const int kb  = i * 64;
        const int cur = i & 1;
        __syncthreads();  // A: previous iteration's reads complete

        if (tid < 64) ldsM[tid] = amask[b * SLEN + kb + tid];
#if HAVE_TDM
        if (wid == 0) {
            if (i + 1 < nkb) {
                const int nxt = cur ^ 1;
                tdm_load_tile_bf16(kbasep + (size_t)(kb + 64) * KQ_LD,
                                   nxt * FA_TBUF, 64, 64, KQ_LD, KQ_LD,
                                   1u << 20);
                tdm_load_tile_bf16(vtbasep + kb + 64,
                                   FA_VTOFF + nxt * FA_TBUF, 64, 64, SLEN,
                                   SLEN, 1u << 20);
                WAIT_TENSORCNT(2);  // tiles(i) landed; tiles(i+1) in flight
            } else {
                WAIT_TENSORCNT(0);
            }
        }
#else
        if (tid < 128) {  // manual vectorized staging (rows are contiguous)
            int row = tid >> 1;
            int c0  = (tid & 1) * 32;
            __bf16* lk = (__bf16*)(smem + cur * FA_TBUF);
            __bf16* lv = (__bf16*)(smem + FA_VTOFF + cur * FA_TBUF);
            const f32x4* ks = reinterpret_cast<const f32x4*>(
                kbasep + (size_t)(kb + row) * KQ_LD + c0);
            const f32x4* vs = reinterpret_cast<const f32x4*>(
                vtbasep + (size_t)row * SLEN + kb + c0);
            f32x4* kd = reinterpret_cast<f32x4*>(&lk[row * 64 + c0]);
            f32x4* vd = reinterpret_cast<f32x4*>(&lv[row * 64 + c0]);
            kd[0] = ks[0]; kd[1] = ks[1]; kd[2] = ks[2]; kd[3] = ks[3];
            vd[0] = vs[0]; vd[1] = vs[1]; vd[2] = vs[2]; vd[3] = vs[3];
        }
#endif
        __syncthreads();  // B: K(cur), Vt(cur), mask ready
        const __bf16* lk = (const __bf16*)(smem + cur * FA_TBUF);
        const __bf16* lv = (const __bf16*)(smem + FA_VTOFF + cur * FA_TBUF);

        // ---- scores: S[16q,64k] = Q K^T ----
        v8f Sacc[4];
#pragma unroll
        for (int t = 0; t < 4; t++) Sacc[t] = zero8();
#pragma unroll
        for (int ks = 0; ks < 2; ks++) {
#pragma unroll
            for (int t = 0; t < 4; t++) {
                v16bf kf = load_b_frag(lk, 64, t * 16, ks * 32, lane);
                Sacc[t] = __builtin_amdgcn_wmma_f32_16x16x32_bf16(
                    false, qf[ks], false, kf, (short)0, Sacc[t], false, false);
            }
        }

        // ---- pre-scale into exp2 domain + key mask ----
#pragma unroll
        for (int t = 0; t < 4; t++) {
            int mv = ldsM[t * 16 + ln];
#pragma unroll
            for (int r = 0; r < 8; r++) {
                float s = Sacc[t][r] * SCALE_LOG2E;
                Sacc[t][r] = (mv == 0) ? -INFINITY : s;
            }
        }

        // ---- online softmax (row reductions within 16-lane half groups) ----
#pragma unroll
        for (int r = 0; r < 8; r++) {
            float rm = fmaxf(fmaxf(Sacc[0][r], Sacc[1][r]),
                             fmaxf(Sacc[2][r], Sacc[3][r]));
            rm = fmaxf(rm, __shfl_xor(rm, 1));
            rm = fmaxf(rm, __shfl_xor(rm, 2));
            rm = fmaxf(rm, __shfl_xor(rm, 4));
            rm = fmaxf(rm, __shfl_xor(rm, 8));
            float mn = fmaxf(mrow[r], rm);
            float sc = (mn == -INFINITY) ? 1.f : exp2f(mrow[r] - mn);
            mrow[r] = mn;
            float rs = 0.f;
#pragma unroll
            for (int t = 0; t < 4; t++) {
                float p = (mn == -INFINITY) ? 0.f : exp2f(Sacc[t][r] - mn);
                Sacc[t][r] = p;
                rs += p;
            }
            rs += __shfl_xor(rs, 1);
            rs += __shfl_xor(rs, 2);
            rs += __shfl_xor(rs, 4);
            rs += __shfl_xor(rs, 8);
            lrow[r] = lrow[r] * sc + rs;
#pragma unroll
            for (int t = 0; t < 4; t++) O[t][r] *= sc;
        }

        // ---- P through per-wave LDS into A-fragment layout ----
        __bf16* lp = ldsP + wid * 16 * 64;
#pragma unroll
        for (int t = 0; t < 4; t++)
#pragma unroll
            for (int r = 0; r < 8; r++)
                lp[(r + 8 * hf) * 64 + t * 16 + ln] = (__bf16)Sacc[t][r];
        // intra-wave LDS ops are in-order; ldsP region is wave-private.

        // ---- O += P V ----
#pragma unroll
        for (int ks = 0; ks < 2; ks++) {
            v16bf pf = load_a_frag(lp, 64, 0, ks * 32, lane);
#pragma unroll
            for (int t = 0; t < 4; t++) {
                v16bf vf = load_b_frag(lv, 64, t * 16, ks * 32, lane);
                O[t] = __builtin_amdgcn_wmma_f32_16x16x32_bf16(
                    false, pf, false, vf, (short)0, O[t], false, false);
            }
        }
    }

    // ---- normalize + store ctx[b*S + q][h*64 + d] as bf16 ----
    __bf16* cb = ctx + (size_t)(b * SLEN + qrow0) * DH + h * HDIM;
#pragma unroll
    for (int r = 0; r < 8; r++) {
        float inv = (lrow[r] > 0.f) ? 1.f / lrow[r] : 0.f;
        int row = r + 8 * hf;
#pragma unroll
        for (int t = 0; t < 4; t++)
            cb[(size_t)row * DH + t * 16 + ln] = (__bf16)(O[t][r] * inv);
    }
}

// ---------------------------------------------------------------------------
// Launcher
// ---------------------------------------------------------------------------
extern "C" void kernel_launch(void* const* d_in, const int* in_sizes, int n_in,
                              void* d_out, int out_size, void* d_ws,
                              size_t ws_size, hipStream_t stream) {
    const float* x     = (const float*)d_in[0];
    const int*   amask = (const int*)d_in[1];
    const float* Wk    = (const float*)d_in[2];
    const float* Wq    = (const float*)d_in[3];
    const float* Wv    = (const float*)d_in[4];
    const float* Aq    = (const float*)d_in[5];
    const float* Bq    = (const float*)d_in[6];
    const float* Av    = (const float*)d_in[7];
    const float* Bv    = (const float*)d_in[8];
    const float* Wp    = (const float*)d_in[9];
    const float* bp    = (const float*)d_in[10];
    float* out = (float*)d_out;

    char* ws = (char*)d_ws;
    __bf16* xb   = (__bf16*)ws; ws += (size_t)MROWS * DM * 2;       // 8 MB
    __bf16* wall = (__bf16*)ws; ws += (size_t)3 * DH * DM * 2;      // 1.5 MB
    __bf16* wpb  = (__bf16*)ws; ws += (size_t)DM * DH * 2;          // 0.5 MB
    __bf16* kqb  = (__bf16*)ws; ws += (size_t)MROWS * KQ_LD * 2;    // 16 MB
    __bf16* vtb  = (__bf16*)ws; ws += (size_t)MROWS * DH * 2;       // 8 MB
    __bf16* ctx  = (__bf16*)ws;                                     // 8 MB

    // Prep: bf16 conversions + LoRA fold into packed weights.
    cvt_f32_to_bf16<<<(MROWS * DM + 255) / 256, 256, 0, stream>>>(x, xb,
                                                                  MROWS * DM);
    build_w_all<<<(3 * DH * DM + 255) / 256, 256, 0, stream>>>(
        Wk, Wq, Wv, Aq, Bq, Av, Bv, wall);
    cvt_f32_to_bf16<<<(DM * DH + 255) / 256, 256, 0, stream>>>(Wp, wpb,
                                                               DM * DH);

    // K|Q projection: [8192,512] x [1024,512]^T -> [8192,1024] bf16.
    dim3 g1(MROWS / 128, KQ_LD / 128);
    gemm_bf16_wmma<0><<<g1, 256, GEMM_LDS, stream>>>(
        xb, wall, kqb, nullptr, nullptr, DM, KQ_LD);

    // V projection with transposed store: -> vt[b*H+h][d][s] bf16.
    dim3 g1v(MROWS / 128, DH / 128);
    gemm_bf16_wmma<2><<<g1v, 256, GEMM_LDS, stream>>>(
        xb, wall + (size_t)2 * DH * DM, vtb, nullptr, nullptr, DM, 0);

    // Flash attention: 128-query blocks x (B*H).
    dim3 g2(SLEN / 128, BATCH * NHEAD);
    flash_attn_wmma<<<g2, 256, FA_LDS, stream>>>(kqb, vtb, amask, ctx);

    // Output projection + bias: [8192,512] x [512,512]^T -> f32 out.
    dim3 g3(MROWS / 128, DM / 128);
    gemm_bf16_wmma<1><<<g3, 256, GEMM_LDS, stream>>>(ctx, wpb, nullptr, out,
                                                     bp, DH, DM);
}
